// GRASP_10264971837873
// MI455X (gfx1250) — matrix operand order repaired and checked by reading, not compiled, not run on hardware
//
#include <hip/hip_runtime.h>

// Problem constants (match reference).
#define NN      100000
#define EE      640000
#define NB      16
#define IND     256
#define HD1     128
#define HD2     64
#define NCLS    2

typedef __attribute__((ext_vector_type(16))) __bf16    v16bf;
typedef __attribute__((ext_vector_type(8)))  __bf16    v8bf;
typedef __attribute__((ext_vector_type(8)))  float     v8f;
typedef __attribute__((ext_vector_type(4)))  unsigned  u32x4;

__device__ __forceinline__ unsigned short f2bf_bits(float f) {
    unsigned u = __builtin_bit_cast(unsigned, f);
    u += 0x7FFFu + ((u >> 16) & 1u);           // round-to-nearest-even
    return (unsigned short)(u >> 16);
}

// ---------------------------------------------------------------- utilities
__global__ void k_zero(float* __restrict__ p, int n) {
    int i = blockIdx.x * blockDim.x + threadIdx.x;
    if (i < n) p[i] = 0.0f;
}

__global__ void k_degrees(const int* __restrict__ src, const int* __restrict__ dst,
                          float* __restrict__ degO, float* __restrict__ degI, int E) {
    int e = blockIdx.x * blockDim.x + threadIdx.x;
    if (e >= E) return;
    atomicAdd(&degO[src[e]], 1.0f);
    atomicAdd(&degI[dst[e]], 1.0f);
}

__global__ void k_rsqrt(const float* __restrict__ degO, const float* __restrict__ degI,
                        float* __restrict__ cSrc, float* __restrict__ cDst, int n) {
    int i = blockIdx.x * blockDim.x + threadIdx.x;
    if (i >= n) return;
    cSrc[i] = rsqrtf(fmaxf(degO[i], 1.0f));
    cDst[i] = rsqrtf(fmaxf(degI[i], 1.0f));
}

// W: [K, dout] fp32 row-major  ->  Wt: [dout, K] bf16 bits (transposed)
__global__ void k_convert_wt(const float* __restrict__ W, unsigned short* __restrict__ Wt,
                             int K, int dout) {
    int i = blockIdx.x * blockDim.x + threadIdx.x;
    int total = K * dout;
    if (i >= total) return;
    int k = i / dout;
    int o = i - k * dout;
    Wt[(size_t)o * K + k] = f2bf_bits(W[i]);
}

// ------------------------------------------------------------ WMMA GEMM
// Y[N,DOUT] = epilogue( X[N,K] @ W[K,DOUT] ), one 16-row tile per block,
// one wave per 16-wide column tile. A tile is converted fp32->bf16 into LDS
// once per block; fragments are built with ds_load_b128 / global_load_b128.
template<int K, int DOUT, bool HAS_SCALE, bool HAS_BIAS, int ACT>
__global__ void k_gemm_wmma(const float* __restrict__ X, const unsigned short* __restrict__ Wt,
                            float* __restrict__ Y, const float* __restrict__ rowScale,
                            const float* __restrict__ bias) {
    constexpr int NTHREADS = (DOUT / 16) * 32;
    __shared__ __align__(16) unsigned short As[16 * K];
    __shared__ float sScale[16];

    const int tid = threadIdx.x;
    const int rowBase = blockIdx.x * 16;

    // ---- stage + convert the 16xK fp32 tile to bf16 in LDS (once per block)
    {
        unsigned* AsU = (unsigned*)As;
        constexpr int PAIRS = 16 * K / 2;
#pragma unroll
        for (int i = tid; i < PAIRS; i += NTHREADS) {
            const int row = i / (K / 2);
            const int kp  = i - row * (K / 2);
            const float2 f2 = *(const float2*)(X + (size_t)(rowBase + row) * K + 2 * kp);
            AsU[i] = (unsigned)f2bf_bits(f2.x) | ((unsigned)f2bf_bits(f2.y) << 16);
        }
        if (HAS_SCALE && tid < 16) sScale[tid] = rowScale[rowBase + tid];
    }
    __syncthreads();

    const int lane = tid & 31;
    const int wave = tid >> 5;
    const int colBase = wave * 16;
    const int l15 = lane & 15;
    const int lhi = lane >> 4;                 // 0 or 1

    const unsigned short* wrow = Wt + (size_t)(colBase + l15) * K;   // B: lane's column
    const unsigned short* arow = As + l15 * K;                       // A: lane's row

    v8f acc = {};
#pragma unroll
    for (int k0 = 0; k0 < K; k0 += 32) {
        // A fragment: half h holds 8 contiguous bf16 at K = k0 + 16h + 8*lhi
        const u32x4 a0 = *(const u32x4*)(arow + k0 + 8 * lhi);
        const u32x4 a1 = *(const u32x4*)(arow + k0 + 16 + 8 * lhi);
        const v8bf alo = __builtin_bit_cast(v8bf, a0);
        const v8bf ahi = __builtin_bit_cast(v8bf, a1);
        const v16bf a = __builtin_shufflevector(alo, ahi,
                            0, 1, 2, 3, 4, 5, 6, 7, 8, 9, 10, 11, 12, 13, 14, 15);
        // B fragment: half h holds 8 contiguous bf16 at K = k0 + 16h + 8*lhi
        const u32x4 b0 = *(const u32x4*)(wrow + k0 + 8 * lhi);
        const u32x4 b1 = *(const u32x4*)(wrow + k0 + 16 + 8 * lhi);
        const v8bf blo = __builtin_bit_cast(v8bf, b0);
        const v8bf bhi = __builtin_bit_cast(v8bf, b1);
        const v16bf b = __builtin_shufflevector(blo, bhi,
                            0, 1, 2, 3, 4, 5, 6, 7, 8, 9, 10, 11, 12, 13, 14, 15);
        acc = __builtin_amdgcn_wmma_f32_16x16x32_bf16(false, a, false, b,
                                                      (short)0, acc, false, false);
    }

    // D layout: lane l, VGPR v -> row = v + 8*lhi, col = l15
    const float bcol = HAS_BIAS ? bias[colBase + l15] : 0.0f;
#pragma unroll
    for (int v = 0; v < 8; ++v) {
        const int row = rowBase + v + 8 * lhi;
        float r = acc[v];
        if constexpr (HAS_SCALE) r *= sScale[v + 8 * lhi];
        if constexpr (HAS_BIAS)  { r += bcol; }
        if constexpr (ACT == 1)  r = fmaxf(r, 0.0f);
        if constexpr (ACT == 2)  r = tanhf(r);
        Y[(size_t)row * DOUT + colBase + l15] = r;
    }
}

// ------------------------------------------------------- message passing
// M[dst[e], :] += Y[src[e], :]   (float4 per thread, f32 atomics)
template<int DLOG2>
__global__ void k_edge_aggregate(const float* __restrict__ Y, const int* __restrict__ src,
                                 const int* __restrict__ dst, float* __restrict__ M, int E) {
    const long long idx = (long long)blockIdx.x * blockDim.x + threadIdx.x;
    constexpr int chunksLog2 = DLOG2 - 2;
    const long long total = (long long)E << chunksLog2;
    if (idx >= total) return;
    const int e = (int)(idx >> chunksLog2);
    const int c = ((int)idx & ((1 << chunksLog2) - 1)) << 2;
    constexpr int d = 1 << DLOG2;
    const float4 val = *(const float4*)(Y + (size_t)src[e] * d + c);
    float* out = M + (size_t)dst[e] * d + c;
    atomicAdd(out + 0, val.x);
    atomicAdd(out + 1, val.y);
    atomicAdd(out + 2, val.z);
    atomicAdd(out + 3, val.w);
}

// X = relu(M * cDst[row] + b[col])
template<int DLOG2>
__global__ void k_post(const float* __restrict__ M, const float* __restrict__ cDst,
                       const float* __restrict__ bias, float* __restrict__ X, int n) {
    const long long idx = (long long)blockIdx.x * blockDim.x + threadIdx.x;
    const long long total = (long long)n << DLOG2;
    if (idx >= total) return;
    const int row = (int)(idx >> DLOG2);
    const int col = (int)idx & ((1 << DLOG2) - 1);
    X[idx] = fmaxf(M[idx] * cDst[row] + bias[col], 0.0f);
}

// ---------------------------------------------------------------- pooling
__global__ void k_pool_sum(const float* __restrict__ X3, const int* __restrict__ gid,
                           float* __restrict__ sums, int n) {
    const int idx = blockIdx.x * blockDim.x + threadIdx.x;
    if (idx >= n * 16) return;
    const int node = idx >> 4;
    const int c = (idx & 15) << 2;
    const int g = gid[node];
    const float4 v = *(const float4*)(X3 + (size_t)node * HD2 + c);
    atomicAdd(&sums[g * HD2 + c + 0], v.x);
    atomicAdd(&sums[g * HD2 + c + 1], v.y);
    atomicAdd(&sums[g * HD2 + c + 2], v.z);
    atomicAdd(&sums[g * HD2 + c + 3], v.w);
}

__global__ void k_pool_cnt(const int* __restrict__ gid, float* __restrict__ cnts, int n) {
    const int i = blockIdx.x * blockDim.x + threadIdx.x;
    if (i < n) atomicAdd(&cnts[gid[i]], 1.0f);
}

// A[n] = dot(T[n,:], aW2) + ab2
__global__ void k_attn_dot(const float* __restrict__ T, const float* __restrict__ aW2,
                           const float* __restrict__ ab2, float* __restrict__ A, int n) {
    const int i = blockIdx.x * blockDim.x + threadIdx.x;
    if (i >= n) return;
    float s = 0.0f;
#pragma unroll
    for (int k = 0; k < HD2; k += 4) {
        const float4 t = *(const float4*)(T + (size_t)i * HD2 + k);
        const float4 w = *(const float4*)(aW2 + k);
        s += t.x * w.x + t.y * w.y + t.z * w.z + t.w * w.w;
    }
    A[i] = s + ab2[0];
}

// out = relu(hg@cW1+cb1)@cW2 + cb2   (single block, 256 threads)
__global__ void k_classifier(const float* __restrict__ sums, const float* __restrict__ cnts,
                             const float* __restrict__ cW1, const float* __restrict__ cb1,
                             const float* __restrict__ cW2, const float* __restrict__ cb2,
                             float* __restrict__ out) {
    __shared__ float hg[NB * HD2];
    __shared__ float z[NB * HD2];
    const int tid = threadIdx.x;
    for (int i = tid; i < NB * HD2; i += 256) {
        const int b = i >> 6;
        hg[i] = sums[i] / fmaxf(cnts[b], 1.0f);
    }
    __syncthreads();
    for (int i = tid; i < NB * HD2; i += 256) {
        const int b = i >> 6, o = i & 63;
        float s = cb1[o];
        for (int k = 0; k < HD2; ++k) s += hg[b * HD2 + k] * cW1[k * HD2 + o];
        z[i] = fmaxf(s, 0.0f);
    }
    __syncthreads();
    if (tid < NB * NCLS) {
        const int b = tid >> 1, c = tid & 1;
        float s = cb2[c];
        for (int k = 0; k < HD2; ++k) s += z[b * HD2 + k] * cW2[k * NCLS + c];
        out[b * NCLS + c] = s;
    }
}

// ---------------------------------------------------------------- driver
extern "C" void kernel_launch(void* const* d_in, const int* in_sizes, int n_in,
                              void* d_out, int out_size, void* d_ws, size_t ws_size,
                              hipStream_t stream) {
    const float* h   = (const float*)d_in[0];
    const int*  src  = (const int*)d_in[1];
    const int*  dst  = (const int*)d_in[2];
    const int*  gid  = (const int*)d_in[3];
    const float* W1  = (const float*)d_in[4];  const float* b1  = (const float*)d_in[5];
    const float* W2  = (const float*)d_in[6];  const float* b2  = (const float*)d_in[7];
    const float* W3  = (const float*)d_in[8];  const float* b3  = (const float*)d_in[9];
    const float* aW1 = (const float*)d_in[10]; const float* ab1 = (const float*)d_in[11];
    const float* aW2 = (const float*)d_in[12]; const float* ab2 = (const float*)d_in[13];
    const float* cW1 = (const float*)d_in[14]; const float* cb1 = (const float*)d_in[15];
    const float* cW2 = (const float*)d_in[16]; const float* cb2 = (const float*)d_in[17];

    float* outC = (float*)d_out;          // [16*2]
    float* outA = outC + NB * NCLS;       // [N]

    // workspace carve (256B aligned)
    char* p = (char*)d_ws;
    auto carve = [&](size_t bytes) -> void* {
        void* r = (void*)p;
        p += (bytes + 255) & ~(size_t)255;
        return r;
    };
    float* degO = (float*)carve((size_t)NN * 4);
    float* degI = (float*)carve((size_t)NN * 4);
    float* cSrc = (float*)carve((size_t)NN * 4);
    float* cDst = (float*)carve((size_t)NN * 4);
    float* bufA = (float*)carve((size_t)NN * HD1 * 4);
    float* bufB = (float*)carve((size_t)NN * HD1 * 4);
    unsigned short* W1t  = (unsigned short*)carve((size_t)IND * HD1 * 2);
    unsigned short* W2t  = (unsigned short*)carve((size_t)HD1 * HD1 * 2);
    unsigned short* W3t  = (unsigned short*)carve((size_t)HD1 * HD2 * 2);
    unsigned short* aW1t = (unsigned short*)carve((size_t)HD2 * HD2 * 2);
    float* sums = (float*)carve((size_t)NB * HD2 * 4);
    float* cnts = (float*)carve((size_t)NB * 4);

    const int T = 256;
    auto blocks = [](long long n, int t) { return (int)((n + t - 1) / t); };

    // degrees -> c_src / c_dst
    k_zero<<<blocks(NN, T), T, 0, stream>>>(degO, NN);
    k_zero<<<blocks(NN, T), T, 0, stream>>>(degI, NN);
    k_degrees<<<blocks(EE, T), T, 0, stream>>>(src, dst, degO, degI, EE);
    k_rsqrt<<<blocks(NN, T), T, 0, stream>>>(degO, degI, cSrc, cDst, NN);

    // bf16 transposed weights
    k_convert_wt<<<blocks(IND * HD1, T), T, 0, stream>>>(W1, W1t, IND, HD1);
    k_convert_wt<<<blocks(HD1 * HD1, T), T, 0, stream>>>(W2, W2t, HD1, HD1);
    k_convert_wt<<<blocks(HD1 * HD2, T), T, 0, stream>>>(W3, W3t, HD1, HD2);
    k_convert_wt<<<blocks(HD2 * HD2, T), T, 0, stream>>>(aW1, aW1t, HD2, HD2);

    const int rowTiles = NN / 16;   // 6250, exact

    // ---- layer 1: h[N,256] -> bufA[N,128] (x * c_src), aggregate, post -> bufA
    k_gemm_wmma<IND, HD1, true, false, 0><<<rowTiles, (HD1 / 16) * 32, 0, stream>>>(h, W1t, bufA, cSrc, nullptr);
    k_zero<<<blocks((long long)NN * HD1, T), T, 0, stream>>>(bufB, NN * HD1);
    k_edge_aggregate<7><<<blocks((long long)EE * (HD1 / 4), T), T, 0, stream>>>(bufA, src, dst, bufB, EE);
    k_post<7><<<blocks((long long)NN * HD1, T), T, 0, stream>>>(bufB, cDst, b1, bufA, NN);

    // ---- layer 2: bufA[N,128] -> bufB, aggregate -> bufA, post -> bufB
    k_gemm_wmma<HD1, HD1, true, false, 0><<<rowTiles, (HD1 / 16) * 32, 0, stream>>>(bufA, W2t, bufB, cSrc, nullptr);
    k_zero<<<blocks((long long)NN * HD1, T), T, 0, stream>>>(bufA, NN * HD1);
    k_edge_aggregate<7><<<blocks((long long)EE * (HD1 / 4), T), T, 0, stream>>>(bufB, src, dst, bufA, EE);
    k_post<7><<<blocks((long long)NN * HD1, T), T, 0, stream>>>(bufA, cDst, b2, bufB, NN);

    // ---- layer 3: bufB[N,128] -> bufA[N,64], aggregate -> bufB, post -> bufA (= X3)
    k_gemm_wmma<HD1, HD2, true, false, 0><<<rowTiles, (HD2 / 16) * 32, 0, stream>>>(bufB, W3t, bufA, cSrc, nullptr);
    k_zero<<<blocks((long long)NN * HD2, T), T, 0, stream>>>(bufB, NN * HD2);
    k_edge_aggregate<6><<<blocks((long long)EE * (HD2 / 4), T), T, 0, stream>>>(bufA, src, dst, bufB, EE);
    k_post<6><<<blocks((long long)NN * HD2, T), T, 0, stream>>>(bufB, cDst, b3, bufA, NN);

    // ---- attention head: T = tanh(X3 @ aW1 + ab1) -> bufB; A = T . aW2 + ab2
    k_gemm_wmma<HD2, HD2, false, true, 2><<<rowTiles, (HD2 / 16) * 32, 0, stream>>>(bufA, aW1t, bufB, nullptr, ab1);
    k_attn_dot<<<blocks(NN, T), T, 0, stream>>>(bufB, aW2, ab2, outA, NN);

    // ---- pooling + classifier
    k_zero<<<blocks(NB * HD2, T), T, 0, stream>>>(sums, NB * HD2);
    k_zero<<<1, T, 0, stream>>>(cnts, NB);
    k_pool_sum<<<blocks((long long)NN * 16, T), T, 0, stream>>>(bufA, gid, sums, NN);
    k_pool_cnt<<<blocks(NN, T), T, 0, stream>>>(gid, cnts, NN);
    k_classifier<<<1, 256, 0, stream>>>(sums, cnts, cW1, cb1, cW2, cb2, outC);
}
